// Qwen2MoeDecoderLayer_4054449127764
// MI455X (gfx1250) — matrix operand order repaired and checked by reading, not compiled, not run on hardware
//
#include <hip/hip_runtime.h>
#include <hip/hip_bf16.h>
#include <cmath>

// ---------------- types ----------------
typedef __bf16 bf16_t;
typedef __attribute__((ext_vector_type(16))) __bf16 v16bf;
typedef __attribute__((ext_vector_type(8)))  float  v8f;

#define TTOK 1024
#define DMODEL 2048
#define NHEAD 16
#define HDIM 128
#define NEXP 16
#define IMOE 1408
#define ISH  5632
#define QKVN (3 * NHEAD * HDIM)   // 6144
#define LDSK 40                   // LDS row pitch (elements): 80B -> 16B-aligned rows

__device__ __forceinline__ float tofl(float x)  { return x; }
__device__ __forceinline__ float tofl(bf16_t x) { return (float)x; }
__device__ __forceinline__ float silu_f(float x) { return x / (1.0f + expf(-x)); }

// ---------------------------------------------------------------------------
// Fragment helpers: ISA 16-bit WMMA layouts (CDNA5 ISA 7.12.2)
//   A 16x32: lanes 0-15 -> M rows; element pair 2j,2j+1 -> K = f(j,g), f(j,g)+1
//   B 32x16: lane -> N col; element pair 2j,2j+1 -> K = f(j,g), f(j,g)+1 (same as A)
// ---------------------------------------------------------------------------
union Frag { v16bf v; unsigned u[8]; };

template<int PITCH>
__device__ __forceinline__ void load_afrag(Frag& f, const bf16_t (*As)[PITCH],
                                           int row, int g)
{
    #pragma unroll
    for (int j = 0; j < 8; ++j) {
        int kk = (j < 4) ? (2 * j + 8 * g) : (16 + 2 * (j - 4) + 8 * g);
        f.u[j] = *(const unsigned*)&As[row][kk];
    }
}

template<int PITCH>
__device__ __forceinline__ void load_bfrag(Frag& f, const bf16_t (*Bs)[PITCH],
                                           int col, int g)
{
    #pragma unroll
    for (int j = 0; j < 8; ++j) {
        int kk = (j < 4) ? (2 * j + 8 * g) : (16 + 2 * (j - 4) + 8 * g);
        f.u[j] = *(const unsigned*)&Bs[col][kk];
    }
}

// ---------------------------------------------------------------------------
// FAST GEMM:  C[M,N](+bias) = A[M,K](bf16) @ B[K,N](TB)
//  - 256 threads = 8 wave32, block tile 128x64, K-step 32
//  - requires: M % 128 == 0, N % 64 == 0, K % 32 == 0 (true for all dense GEMMs)
//  - all global loads are 128-bit; fp32 weights converted to bf16 into LDS
//  - batched over blockIdx.z with element strides (attention heads)
// ---------------------------------------------------------------------------
template<typename TB>
__global__ __launch_bounds__(256) void gemm_wmma_fast(
    const bf16_t* __restrict__ A, int lda, long long abat,
    const TB*     __restrict__ B, int ldb, long long bbat,
    const float*  __restrict__ bias,
    float*        __restrict__ C, bf16_t* __restrict__ Cbf, int ldc, long long cbat,
    int K)
{
    const int mtile = blockIdx.y * 128;
    const int ntile = blockIdx.x * 64;
    const int tid  = threadIdx.x;
    const int lane = tid & 31;
    const int wave = tid >> 5;
    const int wrow = (wave >> 1) * 32;   // 4 row-groups of 32
    const int wcol = (wave & 1) * 32;    // 2 col-groups of 32
    const int g    = lane >> 4;
    const int l16  = lane & 15;

    const long long aoff = (long long)blockIdx.z * abat;
    const long long boff = (long long)blockIdx.z * bbat;

    __shared__ bf16_t As[128][LDSK];
    __shared__ bf16_t Bs[64][LDSK];

    v8f acc[2][2] = {};

    constexpr int EPC   = 16 / sizeof(TB);   // B elements per 16B chunk
    constexpr int CHROW = 64 / EPC;          // chunks per B row

    for (int k0 = 0; k0 < K; k0 += 32) {
        // ---- A tile: 128 rows x 32 bf16, 16B (8-elem) vector loads
        #pragma unroll
        for (int it = 0; it < 2; ++it) {
            int ch = tid + it * 256;         // 512 chunks
            int r  = ch >> 2;
            int c0 = (ch & 3) << 3;
            *(uint4*)&As[r][c0] =
                *(const uint4*)&A[aoff + (long long)(mtile + r) * lda + k0 + c0];
        }
        // ---- B tile: 32 x 64, 16B vector loads, convert->bf16, transpose to Bs[n][k]
        for (int ch = tid; ch < 32 * CHROW; ch += 256) {
            int kr = ch / CHROW;
            int c0 = (ch % CHROW) * EPC;
            uint4 raw = *(const uint4*)&B[boff + (long long)(k0 + kr) * ldb + ntile + c0];
            const TB* el = (const TB*)&raw;
            #pragma unroll
            for (int e = 0; e < EPC; ++e)
                Bs[c0 + e][kr] = (bf16_t)tofl(el[e]);
            if (k0 + 32 < K)                 // gfx1250 global_prefetch of next tile
                __builtin_prefetch(&B[boff + (long long)(k0 + 32 + kr) * ldb + ntile + c0], 0, 0);
        }
        __syncthreads();

        Frag afr[2], bfr[2];
        #pragma unroll
        for (int mt = 0; mt < 2; ++mt) load_afrag<LDSK>(afr[mt], As, wrow + mt * 16 + l16, g);
        #pragma unroll
        for (int nt = 0; nt < 2; ++nt) load_bfrag<LDSK>(bfr[nt], Bs, wcol + nt * 16 + l16, g);
        #pragma unroll
        for (int mt = 0; mt < 2; ++mt)
            #pragma unroll
            for (int nt = 0; nt < 2; ++nt)
                acc[mt][nt] = __builtin_amdgcn_wmma_f32_16x16x32_bf16(
                    false, afr[mt].v, false, bfr[nt].v, (short)0, acc[mt][nt], false, false);
        __syncthreads();
    }

    const long long coff = (long long)blockIdx.z * cbat;
    #pragma unroll
    for (int mt = 0; mt < 2; ++mt) {
        #pragma unroll
        for (int nt = 0; nt < 2; ++nt) {
            int cc = ntile + wcol + nt * 16 + l16;
            float bv = bias ? bias[cc] : 0.0f;
            #pragma unroll
            for (int r = 0; r < 8; ++r) {
                int lr = mtile + wrow + mt * 16 + 8 * g + r;
                float v = acc[mt][nt][r] + bv;
                long long off = coff + (long long)lr * ldc + cc;
                if (C)   C[off]   = v;
                if (Cbf) Cbf[off] = (bf16_t)v;
            }
        }
    }
}

// ---------------------------------------------------------------------------
// MoE grouped GEMM:  C[rows_e, N] = gather(A)[rows_e, K](bf16) @ B[K,N](fp32)
//  - 128 threads = 4 wave32, tile 64x64; M from counts[e] (device side)
//  - A rows gathered via perm list (gate/up) or offset base (down)
//  - gather row/validity hoisted out of the K-loop; invalid rows clamped to
//    a valid row and the payload zeroed branchlessly (v_cndmask, no exec games)
//  - C rows written at device-side base offsets[e] (compacted)
// ---------------------------------------------------------------------------
__global__ __launch_bounds__(128) void gemm_wmma_moe(
    const bf16_t* __restrict__ A, int lda,
    const float*  __restrict__ B, int ldb,
    float*        __restrict__ C, int ldc,
    int N, int K,
    const int* __restrict__ m_eff_ptr,
    const int* __restrict__ a_row_map,       // nullable
    const int* __restrict__ a_row_base_ptr,  // nullable
    const int* __restrict__ c_row_base_ptr)
{
    const int m_eff = *m_eff_ptr;
    const int mtile = blockIdx.y * 64;
    if (mtile >= m_eff) return;              // uniform block-level exit
    const int ntile = blockIdx.x * 64;
    const int tid  = threadIdx.x;
    const int lane = tid & 31;
    const int wave = tid >> 5;
    const int wrow = (wave >> 1) * 32;
    const int wcol = (wave & 1) * 32;
    const int g    = lane >> 4;
    const int l16  = lane & 15;
    const int arb  = a_row_base_ptr ? *a_row_base_ptr : 0;

    __shared__ bf16_t As[64][LDSK];
    __shared__ bf16_t Bs[64][LDSK];

    // ---- loop-invariant gather: per-thread A chunk pointers + validity
    //      (64 rows x 4 chunks of 8 bf16 = 256 chunks; 2 per thread)
    const bf16_t* arow[2];
    bool avalid[2];
    int   ar[2], ac[2];
    #pragma unroll
    for (int it = 0; it < 2; ++it) {
        int ch = tid + it * 128;
        ar[it] = ch >> 2;
        ac[it] = (ch & 3) << 3;
        int lr = mtile + ar[it];
        avalid[it] = lr < m_eff;
        int safe = avalid[it] ? lr : mtile;          // row mtile is always valid here
        int grow = a_row_map ? a_row_map[safe] : (arb + safe);
        arow[it] = A + (long long)grow * lda + ac[it];
    }

    v8f acc[2][2] = {};

    for (int k0 = 0; k0 < K; k0 += 32) {
        // A tile: branchless 16B gathered loads, zero-select for padded rows
        #pragma unroll
        for (int it = 0; it < 2; ++it) {
            uint4 raw = *(const uint4*)(arow[it] + k0);
            if (!avalid[it]) { raw.x = 0u; raw.y = 0u; raw.z = 0u; raw.w = 0u; }
            *(uint4*)&As[ar[it]][ac[it]] = raw;
        }
        // B tile: 32x64 fp32, float4 loads, convert -> transposed bf16
        #pragma unroll
        for (int i = 0; i < 4; ++i) {
            int ch = tid + i * 128;          // 512 float4 chunks
            int kr = ch >> 4;
            int c0 = (ch & 15) << 2;
            uint4 raw = *(const uint4*)&B[(long long)(k0 + kr) * ldb + ntile + c0];
            const float* el = (const float*)&raw;
            #pragma unroll
            for (int e = 0; e < 4; ++e)
                Bs[c0 + e][kr] = (bf16_t)el[e];
            if (k0 + 32 < K)
                __builtin_prefetch(&B[(long long)(k0 + 32 + kr) * ldb + ntile + c0], 0, 0);
        }
        __syncthreads();

        Frag afr[2], bfr[2];
        #pragma unroll
        for (int mt = 0; mt < 2; ++mt) load_afrag<LDSK>(afr[mt], As, wrow + mt * 16 + l16, g);
        #pragma unroll
        for (int nt = 0; nt < 2; ++nt) load_bfrag<LDSK>(bfr[nt], Bs, wcol + nt * 16 + l16, g);
        #pragma unroll
        for (int mt = 0; mt < 2; ++mt)
            #pragma unroll
            for (int nt = 0; nt < 2; ++nt)
                acc[mt][nt] = __builtin_amdgcn_wmma_f32_16x16x32_bf16(
                    false, afr[mt].v, false, bfr[nt].v, (short)0, acc[mt][nt], false, false);
        __syncthreads();
    }

    const int crb = *c_row_base_ptr;
    #pragma unroll
    for (int mt = 0; mt < 2; ++mt) {
        #pragma unroll
        for (int nt = 0; nt < 2; ++nt) {
            int cc = ntile + wcol + nt * 16 + l16;
            #pragma unroll
            for (int r = 0; r < 8; ++r) {
                int lr = mtile + wrow + mt * 16 + 8 * g + r;
                if (lr < m_eff)
                    C[(long long)(crb + lr) * ldc + cc] = acc[mt][nt][r];
            }
        }
    }
}

// ---------------------------------------------------------------------------
// RMSNorm: fp32 row -> bf16 row (x * rsqrt(mean(x^2)+eps) * w)
// ---------------------------------------------------------------------------
__global__ __launch_bounds__(256) void rmsnorm_bf16(const float* __restrict__ x,
                                                    const float* __restrict__ w,
                                                    bf16_t* __restrict__ out)
{
    int t = blockIdx.x, tid = threadIdx.x;
    const float* row = x + (long long)t * DMODEL;
    __shared__ float red[256];
    float s = 0.f;
    for (int d = tid; d < DMODEL; d += 256) { float v = row[d]; s += v * v; }
    red[tid] = s; __syncthreads();
    for (int st = 128; st > 0; st >>= 1) { if (tid < st) red[tid] += red[tid + st]; __syncthreads(); }
    float r = rsqrtf(red[0] / (float)DMODEL + 1e-6f);
    for (int d = tid; d < DMODEL; d += 256)
        out[(long long)t * DMODEL + d] = (bf16_t)(row[d] * r * w[d]);
}

// ---------------------------------------------------------------------------
// RoPE + head-split: qkv fp32 [T,6144] -> q[h][t][hd], kT[h][hd][t], v[h][t][hd] (bf16)
// ---------------------------------------------------------------------------
__global__ void rope_split(const float* __restrict__ qkv, const int* __restrict__ pos,
                           bf16_t* __restrict__ qb, bf16_t* __restrict__ ktb,
                           bf16_t* __restrict__ vb)
{
    int idx = blockIdx.x * blockDim.x + threadIdx.x;     // T*H*64
    if (idx >= TTOK * NHEAD * (HDIM / 2)) return;
    int i = idx & 63;
    int h = (idx >> 6) & (NHEAD - 1);
    int t = idx >> 10;
    float inv = powf(10000.0f, -((2.0f * (float)i) / (float)HDIM));
    float ang = (float)pos[t] * inv;
    float c = cosf(ang), s = sinf(ang);
    const float* base = qkv + (long long)t * QKVN + h * HDIM;
    float q1 = base[i], q2 = base[i + 64];
    long long qo = ((long long)h * TTOK + t) * HDIM;
    qb[qo + i]      = (bf16_t)(q1 * c - q2 * s);
    qb[qo + i + 64] = (bf16_t)(q2 * c + q1 * s);
    const float* kbase = base + DMODEL;
    float k1 = kbase[i], k2 = kbase[i + 64];
    ktb[((long long)h * HDIM + i)      * TTOK + t] = (bf16_t)(k1 * c - k2 * s);
    ktb[((long long)h * HDIM + i + 64) * TTOK + t] = (bf16_t)(k2 * c + k1 * s);
    const float* vbase = base + 2 * DMODEL;
    vb[qo + i]      = (bf16_t)vbase[i];
    vb[qo + i + 64] = (bf16_t)vbase[i + 64];
}

// ---------------------------------------------------------------------------
// Causal softmax row: scores fp32 -> probs bf16 (zeros beyond diagonal)
// ---------------------------------------------------------------------------
__global__ __launch_bounds__(256) void softmax_causal(const float* __restrict__ scores,
                                                      bf16_t* __restrict__ probs,
                                                      float scale)
{
    int t = blockIdx.x, h = blockIdx.y, tid = threadIdx.x;
    const float* row = scores + ((long long)h * TTOK + t) * TTOK;
    bf16_t* prow = probs + ((long long)h * TTOK + t) * TTOK;
    __shared__ float red[256];
    float m = -3.4e38f;
    for (int c = tid; c <= t; c += 256) m = fmaxf(m, row[c] * scale);
    red[tid] = m; __syncthreads();
    for (int st = 128; st > 0; st >>= 1) { if (tid < st) red[tid] = fmaxf(red[tid], red[tid + st]); __syncthreads(); }
    m = red[0]; __syncthreads();
    float s = 0.f;
    for (int c = tid; c <= t; c += 256) s += expf(row[c] * scale - m);
    red[tid] = s; __syncthreads();
    for (int st = 128; st > 0; st >>= 1) { if (tid < st) red[tid] += red[tid + st]; __syncthreads(); }
    float inv = 1.0f / red[0];
    for (int c = tid; c < TTOK; c += 256)
        prow[c] = (bf16_t)((c <= t) ? expf(row[c] * scale - m) * inv : 0.0f);
}

__global__ void add_f32(const float* __restrict__ a, const float* __restrict__ b,
                        float* __restrict__ o, int n)
{
    int i = blockIdx.x * blockDim.x + threadIdx.x;
    if (i < n) o[i] = a[i] + b[i];
}

__global__ void zero16(int* c) { if (threadIdx.x < NEXP) c[threadIdx.x] = 0; }

__global__ void prefix16(const int* __restrict__ c, int* __restrict__ o)
{
    if (threadIdx.x == 0) { int run = 0; for (int e = 0; e < NEXP; ++e) { o[e] = run; run += c[e]; } }
}

// ---------------------------------------------------------------------------
// Router: logits (deterministic tree reduce), softmax16, top-2, slot assignment
// ---------------------------------------------------------------------------
__global__ __launch_bounds__(256) void router_topk(const bf16_t* __restrict__ x2,
                                                   const float* __restrict__ rw,
                                                   int* __restrict__ counts, int* __restrict__ perm,
                                                   int* __restrict__ tok_e, int* __restrict__ tok_slot,
                                                   float* __restrict__ tok_w)
{
    int t = blockIdx.x, tid = threadIdx.x;
    int e = tid & 15, chunk = tid >> 4;                   // 16 chunks x 16 experts
    const bf16_t* xr = x2 + (long long)t * DMODEL;
    float p = 0.f;
    for (int d = chunk * 128; d < chunk * 128 + 128; ++d)
        p += (float)xr[d] * rw[d * NEXP + e];
    __shared__ float red[256];
    red[tid] = p; __syncthreads();
    for (int s = 8; s >= 1; s >>= 1) { if (chunk < s) red[tid] += red[tid + s * 16]; __syncthreads(); }
    if (tid == 0) {
        float mx = -3.4e38f;
        for (int i = 0; i < NEXP; ++i) mx = fmaxf(mx, red[i]);
        float pr[NEXP];
        for (int i = 0; i < NEXP; ++i) pr[i] = expf(red[i] - mx);
        int i0 = 0;
        for (int i = 1; i < NEXP; ++i) if (pr[i] > pr[i0]) i0 = i;
        int i1 = -1; float b1 = -1.f;
        for (int i = 0; i < NEXP; ++i) if (i != i0 && pr[i] > b1) { b1 = pr[i]; i1 = i; }
        float den = pr[i0] + pr[i1];
        int s0 = atomicAdd(&counts[i0], 1);
        int s1 = atomicAdd(&counts[i1], 1);
        perm[i0 * TTOK + s0] = t;
        perm[i1 * TTOK + s1] = t;
        tok_e[2 * t] = i0;  tok_e[2 * t + 1] = i1;
        tok_slot[2 * t] = s0; tok_slot[2 * t + 1] = s1;
        tok_w[2 * t] = pr[i0] / den; tok_w[2 * t + 1] = pr[i1] / den;
    }
}

// ---------------------------------------------------------------------------
// activation fusions
// ---------------------------------------------------------------------------
__global__ void act_moe(const float* __restrict__ gate, const float* __restrict__ up,
                        bf16_t* __restrict__ g, int n)
{
    int i = blockIdx.x * blockDim.x + threadIdx.x;
    if (i < n) g[i] = (bf16_t)(silu_f(gate[i]) * up[i]);
}

__global__ void act_shared(const float* __restrict__ gu, bf16_t* __restrict__ h, int n)
{
    int i = blockIdx.x * blockDim.x + threadIdx.x;
    if (i < n) {
        int t = i / ISH, c = i % ISH;
        const float* row = gu + (long long)t * (2 * ISH);
        h[i] = (bf16_t)(silu_f(row[c]) * row[ISH + c]);
    }
}

__global__ __launch_bounds__(256) void shared_gate(const bf16_t* __restrict__ x2,
                                                   const float* __restrict__ ge,
                                                   float* __restrict__ out)
{
    int t = blockIdx.x, tid = threadIdx.x;
    __shared__ float red[256];
    float s = 0.f;
    for (int d = tid; d < DMODEL; d += 256) s += (float)x2[(long long)t * DMODEL + d] * ge[d];
    red[tid] = s; __syncthreads();
    for (int st = 128; st > 0; st >>= 1) { if (tid < st) red[tid] += red[tid + st]; __syncthreads(); }
    if (tid == 0) out[t] = 1.0f / (1.0f + expf(-red[0]));
}

// ---------------------------------------------------------------------------
// final combine: out = w0*down[row0] + w1*down[row1] + gate*shared
// ---------------------------------------------------------------------------
__global__ void combine_out(const float* __restrict__ down_out,
                            const float* __restrict__ sh_out,
                            const float* __restrict__ sh_g,
                            const int* __restrict__ offs,
                            const int* __restrict__ tok_e, const int* __restrict__ tok_slot,
                            const float* __restrict__ tok_w,
                            float* __restrict__ out, int n)
{
    int i = blockIdx.x * blockDim.x + threadIdx.x;
    if (i >= n) return;
    int t = i / DMODEL, d = i % DMODEL;
    int r0 = offs[tok_e[2 * t]]     + tok_slot[2 * t];
    int r1 = offs[tok_e[2 * t + 1]] + tok_slot[2 * t + 1];
    out[i] = tok_w[2 * t]     * down_out[(long long)r0 * DMODEL + d]
           + tok_w[2 * t + 1] * down_out[(long long)r1 * DMODEL + d]
           + sh_g[t] * sh_out[i];
}

// ---------------------------------------------------------------------------
// host-side launchers
// ---------------------------------------------------------------------------
template<typename TB>
static void launch_fast(hipStream_t s,
                        const bf16_t* A, int lda, long long ab,
                        const TB* B, int ldb, long long bb,
                        const float* bias,
                        float* C, bf16_t* Cbf, int ldc, long long cb,
                        int M, int N, int K, int nbatch)
{
    dim3 grid(N / 64, M / 128, nbatch), block(256);
    gemm_wmma_fast<TB><<<grid, block, 0, s>>>(A, lda, ab, B, ldb, bb, bias,
                                              C, Cbf, ldc, cb, K);
}

static void launch_moe(hipStream_t s,
                       const bf16_t* A, int lda, const float* B, int ldb,
                       float* C, int ldc, int N, int K,
                       const int* meff, const int* amap, const int* abase, const int* cbase)
{
    dim3 grid(N / 64, TTOK / 64, 1), block(128);
    gemm_wmma_moe<<<grid, block, 0, s>>>(A, lda, B, ldb, C, ldc, N, K,
                                         meff, amap, abase, cbase);
}

extern "C" void kernel_launch(void* const* d_in, const int* in_sizes, int n_in,
                              void* d_out, int out_size, void* d_ws, size_t ws_size,
                              hipStream_t stream)
{
    (void)in_sizes; (void)n_in; (void)out_size; (void)ws_size;
    const int*   positions = (const int*)  d_in[0];
    const float* hidden    = (const float*)d_in[1];
    const float* ln1_w     = (const float*)d_in[2];
    const float* ln2_w     = (const float*)d_in[3];
    const float* qkv_w     = (const float*)d_in[4];
    const float* qkv_b     = (const float*)d_in[5];
    const float* o_w       = (const float*)d_in[6];
    const float* router_w  = (const float*)d_in[7];
    const float* w_gate    = (const float*)d_in[8];
    const float* w_up      = (const float*)d_in[9];
    const float* w_down    = (const float*)d_in[10];
    const float* sh_gu_w   = (const float*)d_in[11];
    const float* sh_down_w = (const float*)d_in[12];
    const float* sh_eg     = (const float*)d_in[13];
    float* out = (float*)d_out;            // [0:TD) = hidden_states, [TD:2TD) = residual
    float* residual2 = out + (long long)TTOK * DMODEL;

    // ---- carve workspace (aligned 256B)
    char* p = (char*)d_ws;
    auto carve = [&](size_t bytes) -> void* {
        void* r = (void*)p; p += (bytes + 255) & ~(size_t)255; return r;
    };
    const long long TD = (long long)TTOK * DMODEL;
    bf16_t* x1      = (bf16_t*)carve(TD * 2);
    float*  qkv     = (float*) carve((long long)TTOK * QKVN * 4);
    bf16_t* qb      = (bf16_t*)carve(TD * 2);
    bf16_t* ktb     = (bf16_t*)carve(TD * 2);
    bf16_t* vb      = (bf16_t*)carve(TD * 2);
    float*  scores  = (float*) carve((long long)NHEAD * TTOK * TTOK * 4);
    bf16_t* probs   = (bf16_t*)carve((long long)NHEAD * TTOK * TTOK * 2);
    bf16_t* ctx     = (bf16_t*)carve(TD * 2);
    float*  attn_o  = (float*) carve(TD * 4);
    bf16_t* x2      = (bf16_t*)carve(TD * 2);
    int*    counts  = (int*)   carve(NEXP * 4);
    int*    offs    = (int*)   carve(NEXP * 4);
    int*    perm    = (int*)   carve((long long)NEXP * TTOK * 4);
    int*    tok_e   = (int*)   carve(TTOK * 2 * 4);
    int*    tok_sl  = (int*)   carve(TTOK * 2 * 4);
    float*  tok_w   = (float*) carve(TTOK * 2 * 4);
    float*  gate_o  = (float*) carve((long long)2 * TTOK * IMOE * 4);
    float*  up_o    = (float*) carve((long long)2 * TTOK * IMOE * 4);
    bf16_t* gmoe    = (bf16_t*)carve((long long)2 * TTOK * IMOE * 2);
    float*  down_o  = (float*) carve((long long)2 * TTOK * DMODEL * 4);
    float*  gu      = (float*) carve((long long)TTOK * 2 * ISH * 4);
    bf16_t* hsh     = (bf16_t*)carve((long long)TTOK * ISH * 2);
    float*  sh_out  = (float*) carve(TD * 4);
    float*  sh_g    = (float*) carve(TTOK * 4);

    // ---- 1. ln1 -> x1 (bf16)
    rmsnorm_bf16<<<TTOK, 256, 0, stream>>>(hidden, ln1_w, x1);

    // ---- 2. QKV = x1 @ qkv_w + bias
    launch_fast<float>(stream, x1, DMODEL, 0, qkv_w, QKVN, 0, qkv_b,
                       qkv, nullptr, QKVN, 0, TTOK, QKVN, DMODEL, 1);

    // ---- 3. RoPE + head split
    rope_split<<<(TTOK * NHEAD * (HDIM / 2)) / 256, 256, 0, stream>>>(qkv, positions, qb, ktb, vb);

    // ---- 4. scores[h] = q[h] @ kT[h]   (batched over heads)
    launch_fast<bf16_t>(stream, qb, HDIM, (long long)TTOK * HDIM,
                        ktb, TTOK, (long long)HDIM * TTOK, nullptr,
                        scores, nullptr, TTOK, (long long)TTOK * TTOK,
                        TTOK, TTOK, HDIM, NHEAD);

    // ---- 5. causal softmax -> bf16 probs
    softmax_causal<<<dim3(TTOK, NHEAD), 256, 0, stream>>>(scores, probs, 0.08838834764831845f);

    // ---- 6. ctx[:, h*HD:(h+1)*HD] = probs[h] @ v[h]  (bf16 out, column-batched)
    launch_fast<bf16_t>(stream, probs, TTOK, (long long)TTOK * TTOK,
                        vb, HDIM, (long long)TTOK * HDIM, nullptr,
                        nullptr, ctx, DMODEL, (long long)HDIM,
                        TTOK, HDIM, TTOK, NHEAD);

    // ---- 7. attn_out = ctx @ o_w ; residual2 = hidden + attn_out
    launch_fast<float>(stream, ctx, DMODEL, 0, o_w, DMODEL, 0, nullptr,
                       attn_o, nullptr, DMODEL, 0, TTOK, DMODEL, DMODEL, 1);
    add_f32<<<(int)(TD / 256), 256, 0, stream>>>(hidden, attn_o, residual2, (int)TD);

    // ---- 8. ln2 -> x2 (bf16)
    rmsnorm_bf16<<<TTOK, 256, 0, stream>>>(residual2, ln2_w, x2);

    // ---- 9. router + top-2 (per-expert token lists), prefix offsets
    zero16<<<1, 32, 0, stream>>>(counts);
    router_topk<<<TTOK, 256, 0, stream>>>(x2, router_w, counts, perm, tok_e, tok_sl, tok_w);
    prefix16<<<1, 32, 0, stream>>>(counts, offs);

    // ---- 10. MoE grouped GEMMs (device-side M via counts[e], gather via perm,
    //          compacted output rows via offs[e])
    for (int e = 0; e < NEXP; ++e) {
        launch_moe(stream, x2, DMODEL, w_gate + (long long)e * DMODEL * IMOE, IMOE,
                   gate_o, IMOE, IMOE, DMODEL,
                   counts + e, perm + e * TTOK, nullptr, offs + e);
        launch_moe(stream, x2, DMODEL, w_up + (long long)e * DMODEL * IMOE, IMOE,
                   up_o, IMOE, IMOE, DMODEL,
                   counts + e, perm + e * TTOK, nullptr, offs + e);
    }
    act_moe<<<(2 * TTOK * IMOE) / 256, 256, 0, stream>>>(gate_o, up_o, gmoe, 2 * TTOK * IMOE);
    for (int e = 0; e < NEXP; ++e) {
        launch_moe(stream, gmoe, IMOE, w_down + (long long)e * IMOE * DMODEL, DMODEL,
                   down_o, DMODEL, DMODEL, IMOE,
                   counts + e, nullptr, offs + e, offs + e);
    }

    // ---- 11. shared expert
    launch_fast<float>(stream, x2, DMODEL, 0, sh_gu_w, 2 * ISH, 0, nullptr,
                       gu, nullptr, 2 * ISH, 0, TTOK, 2 * ISH, DMODEL, 1);
    act_shared<<<(TTOK * ISH) / 256, 256, 0, stream>>>(gu, hsh, TTOK * ISH);
    launch_fast<float>(stream, hsh, ISH, 0, sh_down_w, DMODEL, 0, nullptr,
                       sh_out, nullptr, DMODEL, 0, TTOK, DMODEL, ISH, 1);
    shared_gate<<<TTOK, 256, 0, stream>>>(x2, sh_eg, sh_g);

    // ---- 12. combine -> d_out[0:TD)
    combine_out<<<(int)(TD / 256), 256, 0, stream>>>(down_o, sh_out, sh_g, offs,
                                                     tok_e, tok_sl, tok_w, out, (int)TD);
}
